// DecoderBlock_26963804685137
// MI455X (gfx1250) — compile-verified, matrix operand order checked by reading, not run on hardware
//
#include <hip/hip_runtime.h>
#include <hip/hip_bf16.h>
#include <math.h>

typedef __bf16 bf16_t;
typedef __bf16 v16bf __attribute__((ext_vector_type(16)));
typedef __bf16 v8bf  __attribute__((ext_vector_type(8)));
typedef float  v8f   __attribute__((ext_vector_type(8)));
typedef int    v4i_vs __attribute__((vector_size(16)));   // matches builtin param

#define WMMA_BF16(A_, B_, C_) \
  __builtin_amdgcn_wmma_f32_16x16x32_bf16(false, (A_), false, (B_), (short)0, (C_), false, false)

#if defined(__has_builtin)
#if __has_builtin(__builtin_amdgcn_global_load_async_to_lds_b128)
#define USE_ASYNC_LDS 1
#endif
#endif

constexpr int Bc = 2, ND = 2048, NE = 2048, D = 1024, H = 16, HD = 64, DFF = 4096;
constexpr int QTILES = ND / 128;   // 16

// 16-byte global -> LDS copy; async (ASYNCcnt) when available.
__device__ __forceinline__ void stage16(const bf16_t* g, bf16_t* l) {
#ifdef USE_ASYNC_LDS
  __builtin_amdgcn_global_load_async_to_lds_b128(
      (__attribute__((address_space(1))) v4i_vs*)g,
      (__attribute__((address_space(3))) v4i_vs*)l, 0, 0);
#else
  *(uint4*)l = *(const uint4*)g;
#endif
}

__device__ __forceinline__ void stage_wait() {
#ifdef USE_ASYNC_LDS
#if __has_builtin(__builtin_amdgcn_s_wait_asynccnt)
  __builtin_amdgcn_s_wait_asynccnt(0);
#else
  asm volatile("s_wait_asynccnt 0" ::: "memory");
#endif
#endif
}

__device__ __forceinline__ v16bf load_frag(const bf16_t* p) {
  // A/B 16-bit fragment: elements 0..7 at p[0..7] (K=kb..kb+7),
  // elements 8..15 at p[16..23] (K=kb+16..kb+23). Two 16B LDS loads.
  v8bf lo = *(const v8bf*)p;
  v8bf hi = *(const v8bf*)(p + 16);
  v16bf r;
#pragma unroll
  for (int i = 0; i < 8; i++) { r[i] = lo[i]; r[i + 8] = hi[i]; }
  return r;
}

// ---------------------------------------------------------------------------
// fp32 -> bf16 elementwise convert
// ---------------------------------------------------------------------------
__global__ __launch_bounds__(256)
void cvt_kernel(const float* __restrict__ in, bf16_t* __restrict__ out, size_t n) {
  size_t i = (size_t)blockIdx.x * 256 + threadIdx.x;
  if (i < n) out[i] = (bf16_t)in[i];
}

// ---------------------------------------------------------------------------
// LayerNorm (D=1024), fp32 in, bf16 out. One block (256 thr) per row.
// ---------------------------------------------------------------------------
__global__ __launch_bounds__(256)
void ln_kernel(const float* __restrict__ x, const float* __restrict__ g,
               const float* __restrict__ bta, bf16_t* __restrict__ out) {
  const int row = blockIdx.x, tid = threadIdx.x;
  const float* xr = x + (size_t)row * D;
  float v[4], s = 0.f, ss = 0.f;
#pragma unroll
  for (int i = 0; i < 4; i++) { v[i] = xr[tid + i * 256]; s += v[i]; ss += v[i] * v[i]; }
#pragma unroll
  for (int off = 1; off < 32; off <<= 1) {
    s  += __shfl_xor(s,  off, 32);
    ss += __shfl_xor(ss, off, 32);
  }
  __shared__ float rs[8], rss[8];
  __shared__ float mu_s, rv_s;
  const int w = tid >> 5, lane = tid & 31;
  if (lane == 0) { rs[w] = s; rss[w] = ss; }
  __syncthreads();
  if (tid == 0) {
    float S = 0.f, SS = 0.f;
    for (int i = 0; i < 8; i++) { S += rs[i]; SS += rss[i]; }
    float mu  = S * (1.0f / D);
    float var = SS * (1.0f / D) - mu * mu;
    mu_s = mu;
    rv_s = rsqrtf(var + 1e-5f);
  }
  __syncthreads();
  const float mu = mu_s, rv = rv_s;
  bf16_t* orow = out + (size_t)row * D;
#pragma unroll
  for (int i = 0; i < 4; i++) {
    int c = tid + i * 256;
    orow[c] = (bf16_t)((v[i] - mu) * rv * g[c] + bta[c]);
  }
}

// ---------------------------------------------------------------------------
// GEMM: C[M,N] = A[M,K](bf16) * Bw[K,N](bf16) (+bias)(gelu?)(+res) -> f32/bf16
// block tile 128x128, 8 waves (4x2), wave tile 32x64, K-step 32.
// A-tile staged via async global->LDS (no transpose); B-tile via registers
// (transposed on the LDS store). Epilogue is compile-time configured.
// ---------------------------------------------------------------------------
template <bool HAS_BIAS, bool HAS_RES, bool DO_GELU, bool OUT_BF16>
__global__ __launch_bounds__(256)
void gemm_bf16_kernel(const bf16_t* __restrict__ A, const bf16_t* __restrict__ Bw,
                      const float* __restrict__ bias, const float* __restrict__ res,
                      float* __restrict__ outF, bf16_t* __restrict__ outH,
                      int M, int N, int K) {
  __shared__ bf16_t As[128 * 40];   // [m][k], pitch 40
  __shared__ bf16_t Bs[128 * 40];   // [n][k] (transposed on stage), pitch 40

  const int tid = threadIdx.x;
  const int m0 = blockIdx.y * 128, n0 = blockIdx.x * 128;
  const int w = tid >> 5, lane = tid & 31;
  const int wm = w >> 1, wn = w & 1;
  const int lr = lane & 15, hs = lane >> 4;
  const int kb = hs * 8;

  const int r0  = tid >> 2, kc0 = (tid & 3) << 3;    // A staging: 2 rows/thread
  const int kr0 = tid >> 4, nc0 = (tid & 15) << 3;   // B staging: 2 k-rows/thread

  v8f acc[2][4] = {};

  for (int ks = 0; ks < K; ks += 32) {
    // B tile -> registers before the barrier (transpose on LDS store)
    const bf16_t* Bg = Bw + (size_t)(ks + kr0) * N + n0 + nc0;
    uint4 b0 = *(const uint4*)Bg;
    uint4 b1 = *(const uint4*)(Bg + (size_t)16 * N);
    const bf16_t* Ag = A + (size_t)(m0 + r0) * K + ks + kc0;

    // hint next k-step tiles toward the near caches (global_prefetch_b8)
    if (ks + 32 < K) {
      __builtin_prefetch(Ag + 32, 0, 3);
      __builtin_prefetch(Bg + (size_t)32 * N, 0, 3);
    }

    __syncthreads();
    // A tile: direct global->LDS (async engine when available)
    stage16(Ag,                  &As[r0 * 40 + kc0]);
    stage16(Ag + (size_t)64 * K, &As[(r0 + 64) * 40 + kc0]);
    const bf16_t* pb0 = (const bf16_t*)&b0;
    const bf16_t* pb1 = (const bf16_t*)&b1;
#pragma unroll
    for (int i = 0; i < 8; i++) {
      Bs[(nc0 + i) * 40 + kr0]      = pb0[i];
      Bs[(nc0 + i) * 40 + kr0 + 16] = pb1[i];
    }
    stage_wait();
    __syncthreads();

    v16bf af[2], bq[4];
#pragma unroll
    for (int mf = 0; mf < 2; mf++)
      af[mf] = load_frag(&As[(wm * 32 + mf * 16 + lr) * 40 + kb]);
#pragma unroll
    for (int nf = 0; nf < 4; nf++)
      bq[nf] = load_frag(&Bs[(wn * 64 + nf * 16 + lr) * 40 + kb]);
#pragma unroll
    for (int mf = 0; mf < 2; mf++)
#pragma unroll
      for (int nf = 0; nf < 4; nf++)
        acc[mf][nf] = WMMA_BF16(af[mf], bq[nf], acc[mf][nf]);
  }

  // epilogue: C layout — VGPR j: row = j + hs*8; lane lr: col
#pragma unroll
  for (int mf = 0; mf < 2; mf++) {
#pragma unroll
    for (int nf = 0; nf < 4; nf++) {
      const int col = n0 + wn * 64 + nf * 16 + lr;
      float bval = 0.0f;
      if (HAS_BIAS) bval = bias[col];
#pragma unroll
      for (int j = 0; j < 8; j++) {
        const int row = m0 + wm * 32 + mf * 16 + hs * 8 + j;
        float t = acc[mf][nf][j] + bval;
        if (DO_GELU) t = 0.5f * t * (1.0f + erff(t * 0.70710678118654752f));
        if (HAS_RES) t += res[(size_t)row * N + col];
        if (OUT_BF16) outH[(size_t)row * N + col] = (bf16_t)t;
        else          outF[(size_t)row * N + col] = t;
      }
    }
  }
}

// ---------------------------------------------------------------------------
// pack qkv: fp32 [B*ND, 3D] -> q,k,v bf16 [B,H,N,HD]; q pre-scaled by HD^-0.5
// ---------------------------------------------------------------------------
__global__ __launch_bounds__(256)
void pack_qkv_kernel(const float* __restrict__ qkv, bf16_t* __restrict__ q,
                     bf16_t* __restrict__ k, bf16_t* __restrict__ v) {
  size_t i = (size_t)blockIdx.x * 256 + threadIdx.x;
  const size_t n = (size_t)Bc * ND * 3 * D;
  if (i >= n) return;
  int col = (int)(i % (3 * D));
  int row = (int)(i / (3 * D));
  int which = col / D;
  int c = col % D;
  int h = c / HD, dd = c % HD;
  int b = row / ND, nn = row % ND;
  size_t dst = (((size_t)b * H + h) * ND + nn) * HD + dd;
  float val = qkv[i];
  if (which == 0)      q[dst] = (bf16_t)(val * 0.125f);
  else if (which == 1) k[dst] = (bf16_t)val;
  else                 v[dst] = (bf16_t)val;
}

__global__ __launch_bounds__(256)
void pack_heads_kernel(const float* __restrict__ in, bf16_t* __restrict__ out,
                       int Nrows, float scale) {
  size_t i = (size_t)blockIdx.x * 256 + threadIdx.x;
  const size_t n = (size_t)Bc * Nrows * D;
  if (i >= n) return;
  int col = (int)(i % D);
  int row = (int)(i / D);
  int h = col / HD, dd = col % HD;
  int b = row / Nrows, nn = row % Nrows;
  out[(((size_t)b * H + h) * Nrows + nn) * HD + dd] = (bf16_t)(in[i] * scale);
}

// ---------------------------------------------------------------------------
// Flash attention: q,k,v bf16 [B,H,N,HD] (q pre-scaled), out bf16 [B*ND, D].
// Block = (b,h,128-query tile), 8 waves x 16 query rows; 32-key tiles.
// K tile staged async global->LDS; V tile transposed via registers.
// ---------------------------------------------------------------------------
template <bool CAUSAL>
__global__ __launch_bounds__(256)
void attn_kernel(const bf16_t* __restrict__ q, const bf16_t* __restrict__ k,
                 const bf16_t* __restrict__ v, bf16_t* __restrict__ out,
                 int NKEY) {
  __shared__ bf16_t Kl[32 * 72];        // [key][hd], pitch 72
  __shared__ bf16_t Vt[64 * 40];        // [hd][key], pitch 40
  __shared__ bf16_t Pl[8 * 16 * 40];    // per-wave P scratch [row][key]

  const int tid = threadIdx.x;
  const int w = tid >> 5, lane = tid & 31;
  const int lr = lane & 15, hs = lane >> 4, kb = hs * 8;

  const int bz = blockIdx.x;
  const int qt = bz % QTILES;
  const int bh = bz / QTILES;            // b*H + h
  const int hh = bh % H;
  const int b  = bh / H;

  const bf16_t* qbase = q + ((size_t)bh * ND + (size_t)qt * 128) * HD;
  const bf16_t* kbase = k + (size_t)bh * NKEY * HD;
  const bf16_t* vbase = v + (size_t)bh * NKEY * HD;

  // Q fragments (A-layout) kept in registers: rows w*16 + lr, K = hd.
  const bf16_t* qp = qbase + (size_t)(w * 16 + lr) * HD;
  v16bf aq[2];
#pragma unroll
  for (int t = 0; t < 2; t++) aq[t] = load_frag(qp + t * 32 + kb);

  float m_[8], l_[8];
#pragma unroll
  for (int j = 0; j < 8; j++) { m_[j] = -1e30f; l_[j] = 0.0f; }
  v8f Oc[4] = {};

  const int nkt  = CAUSAL ? (qt * 4 + 4) : (NKEY / 32);
  const int key0 = tid >> 3;        // 0..31
  const int hb   = (tid & 7) << 3;  // 0..56

  for (int kt = 0; kt < nkt; ++kt) {
    const size_t goff = ((size_t)kt * 32 + key0) * HD + hb;
    uint4 vvec = *(const uint4*)(vbase + goff);   // V: registers (transpose)
    __syncthreads();                              // prior iter done reading LDS
    stage16(kbase + goff, &Kl[key0 * 72 + hb]);   // K: direct global->LDS
    const bf16_t* pv = (const bf16_t*)&vvec;
#pragma unroll
    for (int i = 0; i < 8; i++) Vt[(hb + i) * 40 + key0] = pv[i];
    stage_wait();
    __syncthreads();

    // S = (q*scale) K^T : 16x32 scores per wave (2 n-frags x 2 k-steps)
    v8f s[2] = {};
#pragma unroll
    for (int nf = 0; nf < 2; nf++) {
#pragma unroll
      for (int t = 0; t < 2; t++) {
        v16bf bk = load_frag(&Kl[(nf * 16 + lr) * 72 + t * 32 + kb]);
        s[nf] = WMMA_BF16(aq[t], bk, s[nf]);
      }
    }

    // online softmax (rows j + hs*8, cols lr and lr+16 per lane)
    const int qg0 = qt * 128 + w * 16 + hs * 8;
#pragma unroll
    for (int j = 0; j < 8; j++) {
      float t0 = s[0][j], t1 = s[1][j];
      if (CAUSAL) {
        int qr = qg0 + j;
        if (kt * 32 + lr      > qr) t0 = -1e30f;
        if (kt * 32 + 16 + lr > qr) t1 = -1e30f;
      }
      float mx = fmaxf(t0, t1);
#pragma unroll
      for (int off = 1; off < 16; off <<= 1) mx = fmaxf(mx, __shfl_xor(mx, off, 32));
      float mn    = fmaxf(m_[j], mx);
      float alpha = __expf(m_[j] - mn);
      float p0 = __expf(t0 - mn);
      float p1 = __expf(t1 - mn);
      float rs = p0 + p1;
#pragma unroll
      for (int off = 1; off < 16; off <<= 1) rs += __shfl_xor(rs, off, 32);
      l_[j] = l_[j] * alpha + rs;
      m_[j] = mn;
#pragma unroll
      for (int nf = 0; nf < 4; nf++) Oc[nf][j] *= alpha;
      s[0][j] = p0; s[1][j] = p1;
    }

    // C-layout -> A-layout via per-wave LDS scratch
    bf16_t* pl = &Pl[w * 640];
#pragma unroll
    for (int j = 0; j < 8; j++) {
      pl[(hs * 8 + j) * 40 + lr]      = (bf16_t)s[0][j];
      pl[(hs * 8 + j) * 40 + 16 + lr] = (bf16_t)s[1][j];
    }
    __syncthreads();

    // O += P(16x32) * V(32x64)
    v16bf ap = load_frag(&pl[lr * 40 + kb]);
#pragma unroll
    for (int nf = 0; nf < 4; nf++) {
      v16bf bv = load_frag(&Vt[(nf * 16 + lr) * 40 + kb]);
      Oc[nf] = WMMA_BF16(ap, bv, Oc[nf]);
    }
  }

  // normalize and write bf16 out[B*ND, D], heads re-merged
  const int qg = qt * 128 + w * 16 + hs * 8;
  bf16_t* ob = out + (size_t)b * ND * D + (size_t)hh * HD;
#pragma unroll
  for (int j = 0; j < 8; j++) {
    float inv = 1.0f / l_[j];
    size_t rowoff = (size_t)(qg + j) * D;
#pragma unroll
    for (int nf = 0; nf < 4; nf++)
      ob[rowoff + nf * 16 + lr] = (bf16_t)(Oc[nf][j] * inv);
  }
}

// ---------------------------------------------------------------------------
extern "C" void kernel_launch(void* const* d_in, const int* in_sizes, int n_in,
                              void* d_out, int out_size, void* d_ws, size_t ws_size,
                              hipStream_t stream) {
  const float* x     = (const float*)d_in[0];
  const float* memin = (const float*)d_in[1];
  const float* ln1_g = (const float*)d_in[2];
  const float* ln1_b = (const float*)d_in[3];
  const float* Wqkv  = (const float*)d_in[4];
  const float* Wo_sa = (const float*)d_in[5];
  const float* bo_sa = (const float*)d_in[6];
  const float* ln2_g = (const float*)d_in[7];
  const float* ln2_b = (const float*)d_in[8];
  const float* Wq    = (const float*)d_in[9];
  const float* Wk    = (const float*)d_in[10];
  const float* Wv    = (const float*)d_in[11];
  const float* Wo_ca = (const float*)d_in[12];
  const float* bo_ca = (const float*)d_in[13];
  const float* ln3_g = (const float*)d_in[14];
  const float* ln3_b = (const float*)d_in[15];
  const float* W1    = (const float*)d_in[16];
  const float* b1    = (const float*)d_in[17];
  const float* W2    = (const float*)d_in[18];
  const float* b2    = (const float*)d_in[19];

  const int M = Bc * ND;  // 4096 (also == Bc*NE)

  char* p = (char*)d_ws;
  auto alloc = [&](size_t bytes) -> void* {
    char* r = p;
    p += (bytes + 255) & ~(size_t)255;
    return (void*)r;
  };

  bf16_t* wqkv_h = (bf16_t*)alloc((size_t)D * 3 * D * 2);
  bf16_t* wosa_h = (bf16_t*)alloc((size_t)D * D * 2);
  bf16_t* wq_h   = (bf16_t*)alloc((size_t)D * D * 2);
  bf16_t* wk_h   = (bf16_t*)alloc((size_t)D * D * 2);
  bf16_t* wv_h   = (bf16_t*)alloc((size_t)D * D * 2);
  bf16_t* woca_h = (bf16_t*)alloc((size_t)D * D * 2);
  bf16_t* w1_h   = (bf16_t*)alloc((size_t)D * DFF * 2);
  bf16_t* w2_h   = (bf16_t*)alloc((size_t)DFF * D * 2);
  bf16_t* mem_h  = (bf16_t*)alloc((size_t)M * D * 2);
  bf16_t* h_h    = (bf16_t*)alloc((size_t)M * D * 2);
  float*  gemm_f = (float*) alloc((size_t)M * 3 * D * 4);
  bf16_t* q_h    = (bf16_t*)alloc((size_t)M * D * 2);
  bf16_t* k_h    = (bf16_t*)alloc((size_t)M * D * 2);
  bf16_t* v_h    = (bf16_t*)alloc((size_t)M * D * 2);
  bf16_t* ao_h   = (bf16_t*)alloc((size_t)M * D * 2);
  float*  x1     = (float*) alloc((size_t)M * D * 4);
  float*  x2     = (float*) alloc((size_t)M * D * 4);
  bf16_t* ffn_h  = (bf16_t*)alloc((size_t)M * DFF * 2);

  auto cvt = [&](const float* s, bf16_t* dptr, size_t n) {
    cvt_kernel<<<(unsigned)((n + 255) / 256), 256, 0, stream>>>(s, dptr, n);
  };
  cvt(Wqkv,  wqkv_h, (size_t)D * 3 * D);
  cvt(Wo_sa, wosa_h, (size_t)D * D);
  cvt(Wq,    wq_h,   (size_t)D * D);
  cvt(Wk,    wk_h,   (size_t)D * D);
  cvt(Wv,    wv_h,   (size_t)D * D);
  cvt(Wo_ca, woca_h, (size_t)D * D);
  cvt(W1,    w1_h,   (size_t)D * DFF);
  cvt(W2,    w2_h,   (size_t)DFF * D);
  cvt(memin, mem_h,  (size_t)M * D);

  // ---- masked self-attention ----
  ln_kernel<<<M, 256, 0, stream>>>(x, ln1_g, ln1_b, h_h);
  gemm_bf16_kernel<false, false, false, false>
      <<<dim3(3 * D / 128, M / 128), 256, 0, stream>>>(
          h_h, wqkv_h, nullptr, nullptr, gemm_f, nullptr, M, 3 * D, D);
  pack_qkv_kernel<<<(unsigned)(((size_t)M * 3 * D + 255) / 256), 256, 0, stream>>>(
      gemm_f, q_h, k_h, v_h);
  attn_kernel<true><<<Bc * H * QTILES, 256, 0, stream>>>(q_h, k_h, v_h, ao_h, ND);
  gemm_bf16_kernel<true, true, false, false>
      <<<dim3(D / 128, M / 128), 256, 0, stream>>>(
          ao_h, wosa_h, bo_sa, x, x1, nullptr, M, D, D);

  // ---- cross-attention ----
  ln_kernel<<<M, 256, 0, stream>>>(x1, ln2_g, ln2_b, h_h);
  float* q2f = gemm_f;
  float* k2f = gemm_f + (size_t)M * D;
  float* v2f = gemm_f + (size_t)2 * M * D;
  gemm_bf16_kernel<false, false, false, false>
      <<<dim3(D / 128, M / 128), 256, 0, stream>>>(
          h_h, wq_h, nullptr, nullptr, q2f, nullptr, M, D, D);
  gemm_bf16_kernel<false, false, false, false>
      <<<dim3(D / 128, M / 128), 256, 0, stream>>>(
          mem_h, wk_h, nullptr, nullptr, k2f, nullptr, M, D, D);
  gemm_bf16_kernel<false, false, false, false>
      <<<dim3(D / 128, M / 128), 256, 0, stream>>>(
          mem_h, wv_h, nullptr, nullptr, v2f, nullptr, M, D, D);
  unsigned pk = (unsigned)(((size_t)M * D + 255) / 256);
  pack_heads_kernel<<<pk, 256, 0, stream>>>(q2f, q_h, ND, 0.125f);
  pack_heads_kernel<<<pk, 256, 0, stream>>>(k2f, k_h, NE, 1.0f);
  pack_heads_kernel<<<pk, 256, 0, stream>>>(v2f, v_h, NE, 1.0f);
  attn_kernel<false><<<Bc * H * QTILES, 256, 0, stream>>>(q_h, k_h, v_h, ao_h, NE);
  gemm_bf16_kernel<true, true, false, false>
      <<<dim3(D / 128, M / 128), 256, 0, stream>>>(
          ao_h, woca_h, bo_ca, x1, x2, nullptr, M, D, D);

  // ---- FFN ----
  ln_kernel<<<M, 256, 0, stream>>>(x2, ln3_g, ln3_b, h_h);
  gemm_bf16_kernel<true, false, true, true>
      <<<dim3(DFF / 128, M / 128), 256, 0, stream>>>(
          h_h, w1_h, b1, nullptr, nullptr, ffn_h, M, DFF, D);
  gemm_bf16_kernel<true, true, false, false>
      <<<dim3(D / 128, M / 128), 256, 0, stream>>>(
          ffn_h, w2_h, b2, x2, (float*)d_out, nullptr, M, D, DFF);
}
